// SparseArch_87101936762948
// MI455X (gfx1250) — matrix operand order repaired and checked by reading, not compiled, not run on hardware
//
#include <hip/hip_runtime.h>

// Problem constants (match reference)
#define F_    16
#define B_    32768
#define E_    64
#define O_    64
#define HID_  128
#define HASH_ 200000

typedef __attribute__((ext_vector_type(16))) __bf16 v16bf;
typedef __attribute__((ext_vector_type(8)))  float  v8f;

union Frag {
  v16bf v;
  unsigned int u[8];
};

// Native f32 -> bf16 conversion (RNE); lowers to hardware cvt on gfx1250
__device__ __forceinline__ unsigned short f2bf(float a) {
  __bf16 h = (__bf16)a;
  return __builtin_bit_cast(unsigned short, h);
}

__device__ __forceinline__ unsigned int pack2_bf16(float a, float b) {
  return (unsigned int)f2bf(a) | ((unsigned int)f2bf(b) << 16);
}

// A-fragment (16x32 bf16): lane holds row m, K split per ISA layout:
//   VGPR v (v<4): K = k0 + 2v + 8*(lane>>4);  (v>=4): +8 more.
__device__ __forceinline__ void load_A(const unsigned short* base, int stride,
                                       int m, int k0, int lane, Frag& fr) {
  const unsigned short* p = base + (size_t)m * stride + k0 + ((lane >> 4) << 3);
  #pragma unroll
  for (int v = 0; v < 8; ++v) {
    int k = 2 * v + ((v & 4) << 1);
    fr.u[v] = *reinterpret_cast<const unsigned int*>(p + k);
  }
}

// B-fragment (32x16 bf16) from a K-contiguous (transposed) weight row:
//   VGPR v (v<4): K = k0 + 2v + 8*(lane>>4);  (v>=4): +8 more.
__device__ __forceinline__ void load_B(const unsigned short* base, int stride,
                                       int n, int k0, int lane, Frag& fr) {
  const unsigned short* p = base + (size_t)n * stride + k0 + ((lane >> 4) << 3);
  #pragma unroll
  for (int v = 0; v < 8; ++v)
    fr.u[v] = *reinterpret_cast<const unsigned int*>(p + 2 * v + ((v & 4) << 1));
}

__global__ void __launch_bounds__(256)
sparse_arch_kernel(const int*   __restrict__ inp,     // [B, F] indices
                   const float* __restrict__ tables,  // [F, HASH, E]
                   const float* __restrict__ W1,      // [F, E, HID]
                   const float* __restrict__ b1,      // [F, HID]
                   const float* __restrict__ W2,      // [F, HID, O]
                   const float* __restrict__ b2,      // [F, O]
                   float*       __restrict__ out) {   // [F, B, O]
  __shared__ __align__(16) unsigned short sEMB[64][E_];     // 8 KB  (bf16)
  __shared__ __align__(16) unsigned short sW1T[HID_][E_];   // 16 KB (W1 transposed, bf16)
  __shared__ __align__(16) unsigned short sW2T[O_][HID_];   // 16 KB (W2 transposed, bf16)
  __shared__ __align__(16) unsigned short sH[64][HID_];     // 16 KB (hidden, bf16)
  __shared__ float sB1[HID_];
  __shared__ float sB2[O_];

  const int t    = threadIdx.x;
  const int lane = t & 31;
  const int w    = t >> 5;               // wave id 0..7
  const int f    = blockIdx.y;           // feature
  const int tile0 = blockIdx.x * 8;      // 8 batch tiles of 64 rows per block

  // ---------------- stage 0: weights + biases -> LDS (bf16, transposed) ----
  {
    const float* W1f = W1 + (size_t)f * E_ * HID_;   // [e][h], h contiguous
    const int e  = t >> 2;                           // 0..63
    const int h0 = (t & 3) * 32;                     // 0,32,64,96
    #pragma unroll
    for (int j = 0; j < 8; ++j) {
      float4 v = *reinterpret_cast<const float4*>(W1f + (size_t)e * HID_ + h0 + 4 * j);
      sW1T[h0 + 4 * j + 0][e] = f2bf(v.x);
      sW1T[h0 + 4 * j + 1][e] = f2bf(v.y);
      sW1T[h0 + 4 * j + 2][e] = f2bf(v.z);
      sW1T[h0 + 4 * j + 3][e] = f2bf(v.w);
    }
    const float* W2f = W2 + (size_t)f * HID_ * O_;   // [h][o], o contiguous
    const int h  = t >> 1;                           // 0..127
    const int o0 = (t & 1) * 32;                     // 0,32
    #pragma unroll
    for (int j = 0; j < 8; ++j) {
      float4 v = *reinterpret_cast<const float4*>(W2f + (size_t)h * O_ + o0 + 4 * j);
      sW2T[o0 + 4 * j + 0][h] = f2bf(v.x);
      sW2T[o0 + 4 * j + 1][h] = f2bf(v.y);
      sW2T[o0 + 4 * j + 2][h] = f2bf(v.z);
      sW2T[o0 + 4 * j + 3][h] = f2bf(v.w);
    }
    if (t < HID_) sB1[t] = b1[(size_t)f * HID_ + t];
    if (t < O_)   sB2[t] = b2[(size_t)f * O_ + t];
  }

  for (int it = 0; it < 8; ++it) {
    const int bRow0 = (tile0 + it) * 64;

    // -------------- gather 64 embedding rows -> LDS bf16 ------------------
    {
      const int row = t >> 2;          // 0..63
      const int seg = (t & 3) * 16;    // 16 floats per thread
      int gi = inp[(size_t)(bRow0 + row) * F_ + f];
      unsigned int idx = (unsigned int)(gi + 1) % HASH_;
      const float* rp = tables + ((size_t)f * HASH_ + idx) * E_ + seg;
      #pragma unroll
      for (int j = 0; j < 4; ++j) {
        float4 v = *reinterpret_cast<const float4*>(rp + 4 * j);
        uint2 pk;
        pk.x = pack2_bf16(v.x, v.y);
        pk.y = pack2_bf16(v.z, v.w);
        *reinterpret_cast<uint2*>(&sEMB[row][seg + 4 * j]) = pk;
      }
      // software prefetch of next tile's row into the cache hierarchy
      if (it + 1 < 8) {
        int gi2 = inp[(size_t)(bRow0 + 64 + row) * F_ + f];
        unsigned int idx2 = (unsigned int)(gi2 + 1) % HASH_;
        __builtin_prefetch(tables + ((size_t)f * HASH_ + idx2) * E_ + seg, 0, 1);
      }
    }
    __syncthreads();

    // -------------- GEMM1: H(64x128) = relu(EMB(64x64) x W1(64x128) + b1) --
    {
      const int n = (w << 4) + (lane & 15);   // wave w owns HID col tile w
      Frag bf0, bf1;
      load_B(&sW1T[0][0], E_, n, 0,  lane, bf0);
      load_B(&sW1T[0][0], E_, n, 32, lane, bf1);
      const float bias = sB1[n];
      #pragma unroll
      for (int rt = 0; rt < 4; ++rt) {
        Frag a0, a1;
        load_A(&sEMB[0][0], E_, (rt << 4) + (lane & 15), 0,  lane, a0);
        load_A(&sEMB[0][0], E_, (rt << 4) + (lane & 15), 32, lane, a1);
        v8f acc = {};
        acc = __builtin_amdgcn_wmma_f32_16x16x32_bf16(false, a0.v, false, bf0.v,
                                                      (short)0, acc, false, false);
        acc = __builtin_amdgcn_wmma_f32_16x16x32_bf16(false, a1.v, false, bf1.v,
                                                      (short)0, acc, false, false);
        const int mb = (rt << 4) + ((lane >> 4) << 3);
        #pragma unroll
        for (int r = 0; r < 8; ++r) {
          float hv = acc[r] + bias;
          hv = fmaxf(hv, 0.0f);
          sH[mb + r][n] = f2bf(hv);
        }
      }
    }
    __syncthreads();

    // -------------- GEMM2: OUT(64x64) = H(64x128) x W2(128x64) + b2 -------
    {
      const int n = ((w & 3) << 4) + (lane & 15);   // output col tile = w&3
      Frag bw[4];
      #pragma unroll
      for (int kc = 0; kc < 4; ++kc)
        load_B(&sW2T[0][0], HID_, n, kc * 32, lane, bw[kc]);
      const float bias = sB2[n];
      #pragma unroll
      for (int i = 0; i < 2; ++i) {
        const int rt = (w >> 2) + (i << 1);         // row tiles {0,2} or {1,3}
        v8f acc = {};
        #pragma unroll
        for (int kc = 0; kc < 4; ++kc) {
          Frag a;
          load_A(&sH[0][0], HID_, (rt << 4) + (lane & 15), kc * 32, lane, a);
          acc = __builtin_amdgcn_wmma_f32_16x16x32_bf16(false, a.v, false, bw[kc].v,
                                                        (short)0, acc, false, false);
        }
        const int mb = (rt << 4) + ((lane >> 4) << 3);
        float* op = out + ((size_t)f * B_ + bRow0 + mb) * O_ + n;
        #pragma unroll
        for (int r = 0; r < 8; ++r)
          op[(size_t)r * O_] = acc[r] + bias;
      }
    }
    __syncthreads();   // protect sEMB/sH reuse next iteration
  }
}

extern "C" void kernel_launch(void* const* d_in, const int* in_sizes, int n_in,
                              void* d_out, int out_size, void* d_ws, size_t ws_size,
                              hipStream_t stream) {
  const int*   inputs = (const int*)  d_in[0];
  const float* tables = (const float*)d_in[1];
  const float* W1     = (const float*)d_in[2];
  const float* b1     = (const float*)d_in[3];
  const float* W2     = (const float*)d_in[4];
  const float* b2     = (const float*)d_in[5];
  float* out = (float*)d_out;

  dim3 grid(B_ / (64 * 8), F_);   // (64, 16) blocks, 8 batch tiles each
  sparse_arch_kernel<<<grid, 256, 0, stream>>>(inputs, tables, W1, b1, W2, b2, out);
}
